// GraphAttentionCell_72378788872762
// MI455X (gfx1250) — compile-verified, matrix operand order checked by reading, not run on hardware
//
#include <hip/hip_runtime.h>
#include <hip/hip_bf16.h>

typedef __attribute__((ext_vector_type(16))) _Float16 v16h;
typedef __attribute__((ext_vector_type(8)))  _Float16 v8h;
typedef __attribute__((ext_vector_type(8)))  float    v8f;
typedef __attribute__((ext_vector_type(4)))  float    v4f;
typedef __attribute__((ext_vector_type(2)))  float    v2f;

#define B_ 8
#define N_ 512
#define F_ 128
#define H_ 4
#define C_ 64
#define NEG_BIG (-1e10f)
#define LDS_STRIDE 516   // even (b64-aligned) + 4m bank pattern -> conflict-free

// ---------------------------------------------------------------------------
// prep: X -> f16 (row major), kernels[H,F,C] -> f16 transposed Kt[H,C,F]
// ---------------------------------------------------------------------------
__global__ void k_convert_x(const float* __restrict__ X, _Float16* __restrict__ Xh) {
    int idx = blockIdx.x * blockDim.x + threadIdx.x;
    if (idx < B_ * N_ * F_) Xh[idx] = (_Float16)X[idx];
}

__global__ void k_convert_w(const float* __restrict__ K, _Float16* __restrict__ Kt) {
    int idx = blockIdx.x * blockDim.x + threadIdx.x;
    if (idx < H_ * F_ * C_) {
        int h   = idx / (F_ * C_);
        int rem = idx % (F_ * C_);
        int f   = rem / C_;
        int c   = rem % C_;
        Kt[(h * C_ + c) * F_ + f] = (_Float16)K[idx];
    }
}

// ---------------------------------------------------------------------------
// feats via f16 WMMA (f32 accum). Stores:
//   featsT[bh][c][n]  (transposed, for coalesced B-fragments + logit kernel)
//   out[b,n, h*C+c] = relu(feats)   (emb half of output)
// block: 256 thr = 8 waves; each wave one 16x16 tile; block = 32 rows x 64 cols
// ---------------------------------------------------------------------------
__global__ __launch_bounds__(256)
void k_feats(const _Float16* __restrict__ Xh, const _Float16* __restrict__ Kt,
             float* __restrict__ featsT, float* __restrict__ out) {
    const int bh   = blockIdx.x >> 4;   // B*H = 32
    const int rblk = blockIdx.x & 15;   // 16 blocks of 32 rows
    const int b = bh >> 2, h = bh & 3;
    const int wave = threadIdx.x >> 5;
    const int lane = threadIdx.x & 31;
    const int rowbase = rblk * 32 + (wave >> 2) * 16;
    const int colbase = (wave & 3) * 16;
    const int m     = lane & 15;            // A row / B column within tile
    const int khalf = (lane >> 4) * 8;      // 16-bit WMMA K sub-offset per lane half

    const _Float16* aptr = Xh + (size_t)(b * N_ + rowbase + m) * F_ + khalf;
    const _Float16* bptr = Kt + (size_t)(h * C_ + colbase + m) * F_ + khalf;

    v8f acc = {};
#pragma unroll
    for (int k0 = 0; k0 < F_; k0 += 32) {
        v8h alo = *(const v8h*)(aptr + k0);
        v8h ahi = *(const v8h*)(aptr + k0 + 16);
        v8h blo = *(const v8h*)(bptr + k0);
        v8h bhi = *(const v8h*)(bptr + k0 + 16);
        v16h a, bm;
#pragma unroll
        for (int t = 0; t < 8; ++t) {
            a[t] = alo[t]; a[8 + t] = ahi[t];
            bm[t] = blo[t]; bm[8 + t] = bhi[t];
        }
        acc = __builtin_amdgcn_wmma_f32_16x16x32_f16(
            false, a, false, bm, (short)0, acc, false, false);
    }

    // D layout: lane half selects rows v / v+8, col = lane&15
    const int col   = colbase + (lane & 15);
    const int rhalf = (lane >> 4) * 8;
    const int row0  = rowbase + rhalf;

    // transposed feats: 8 consecutive n for fixed c -> two b128 stores
    float* ft = featsT + (size_t)(bh * C_ + col) * N_ + row0;
    v4f lo, hi;
#pragma unroll
    for (int t = 0; t < 4; ++t) { lo[t] = acc[t]; hi[t] = acc[4 + t]; }
    *(v4f*)(ft)     = lo;
    *(v4f*)(ft + 4) = hi;

#pragma unroll
    for (int v = 0; v < 8; ++v) {
        out[(size_t)(b * N_ + row0 + v) * (2 * H_ * C_) + h * C_ + col] =
            fmaxf(acc[v], 0.0f);
    }
}

// ---------------------------------------------------------------------------
// a_s[bh,n] = sum_c featsT[bh,c,n]*attn_self[h,c]; a_n likewise.
// Thread per node: every load coalesced along n.
// ---------------------------------------------------------------------------
__global__ __launch_bounds__(256)
void k_logit_vec(const float* __restrict__ featsT, const float* __restrict__ a_self,
                 const float* __restrict__ a_neigh,
                 float* __restrict__ as_out, float* __restrict__ an_out) {
    const int idx = blockIdx.x * blockDim.x + threadIdx.x;   // [0, B*H*N)
    const int bh = idx / N_;
    const int n  = idx % N_;
    const int h  = bh & (H_ - 1);
    const float* fc = featsT + (size_t)bh * C_ * N_ + n;
    float s = 0.0f, t = 0.0f;
#pragma unroll
    for (int c = 0; c < C_; ++c) {
        float f = fc[(size_t)c * N_];
        s += f * a_self[h * C_ + c];
        t += f * a_neigh[h * C_ + c];
    }
    as_out[idx] = s;
    an_out[idx] = t;
}

// ---------------------------------------------------------------------------
// per (b,h, 16-row tile): masked-softmax 16x512 in LDS (vectorized v4f), then
// out_h = attn @ feats via f32 WMMA 16x16x4 (K=512), relu(+bias) into output.
// block: 128 thr = 4 waves; wave w -> c-tile [16w,16w+16)
// ---------------------------------------------------------------------------
__global__ __launch_bounds__(128)
void k_attn(const float* __restrict__ A, const float* __restrict__ featsT,
            const float* __restrict__ a_s, const float* __restrict__ a_n,
            const float* __restrict__ biases, float* __restrict__ out) {
    __shared__ float attn_s[16][LDS_STRIDE];
    const int bh = blockIdx.x >> 5;          // 32 i-tiles per (b,h)
    const int itile = blockIdx.x & 31;
    const int b = bh >> 2, h = bh & 3;
    const int i0 = itile * 16;
    const int wave = threadIdx.x >> 5, lane = threadIdx.x & 31;

    const float* an_row = a_n + (size_t)bh * N_;

    // ---- phase 1: logits + softmax, 4 rows per wave, v4f throughout ----
    for (int q = 0; q < 4; ++q) {
        const int r = wave * 4 + q;
        const int i = i0 + r;
        const float as = a_s[(size_t)bh * N_ + i];
        const float* arow = A + ((size_t)b * N_ + i) * N_;
        float mx = -3.4e38f;
#pragma unroll
        for (int jv = lane; jv < N_ / 4; jv += 32) {
            const int j = jv * 4;
            v4f av  = *(const v4f*)(arow + j);
            v4f anv = *(const v4f*)(an_row + j);
            v4f e;
#pragma unroll
            for (int t = 0; t < 4; ++t) {
                float x = as + anv[t];
                x = (x > 0.0f) ? x : 0.2f * x;        // LeakyReLU(0.2)
                x += NEG_BIG * (1.0f - av[t]);        // mask non-edges
                e[t] = x;
                mx = fmaxf(mx, x);
            }
            *(v4f*)&attn_s[r][j] = e;
        }
#pragma unroll
        for (int mk = 16; mk >= 1; mk >>= 1) mx = fmaxf(mx, __shfl_xor(mx, mk, 32));
        float sum = 0.0f;
#pragma unroll
        for (int jv = lane; jv < N_ / 4; jv += 32) {
            const int j = jv * 4;
            v4f e = *(const v4f*)&attn_s[r][j];
            v4f p;
#pragma unroll
            for (int t = 0; t < 4; ++t) { p[t] = __expf(e[t] - mx); sum += p[t]; }
            *(v4f*)&attn_s[r][j] = p;
        }
#pragma unroll
        for (int mk = 16; mk >= 1; mk >>= 1) sum += __shfl_xor(sum, mk, 32);
        const float inv = 1.0f / sum;
#pragma unroll
        for (int jv = lane; jv < N_ / 4; jv += 32) {
            const int j = jv * 4;
            v4f p = *(const v4f*)&attn_s[r][j];
#pragma unroll
            for (int t = 0; t < 4; ++t) p[t] *= inv;
            *(v4f*)&attn_s[r][j] = p;
        }
    }
    __syncthreads();

    // ---- phase 2: 16x16 tile of attn(16x512) @ feats(512x16), f32 WMMA ----
    const int c0   = wave * 16;
    const int mrow = lane & 15;               // A row / B column within tile
    const int kk   = (lane >> 4) * 2;         // f32 WMMA K sub-offset per lane half
    const float* fb = featsT + (size_t)(bh * C_ + c0 + mrow) * N_;  // column = contiguous n

    v8f acc = {};
#pragma unroll 4
    for (int j0 = 0; j0 < N_; j0 += 4) {
        v2f a  = *(const v2f*)&attn_s[mrow][j0 + kk];   // one b64 LDS read
        v2f bm = *(const v2f*)(fb + j0 + kk);           // one b64 global read
        acc = __builtin_amdgcn_wmma_f32_16x16x4_f32(
            false, a, false, bm, (short)0, acc, false, false);
    }

    const int col   = lane & 15;
    const int rhalf = (lane >> 4) * 8;
    const float bias = biases[h * C_ + c0 + col];
#pragma unroll
    for (int v = 0; v < 8; ++v) {
        int row = i0 + rhalf + v;
        float val = acc[v] + bias;
        // relu(agg) == relu(output) (maxpool eliminated algebraically)
        out[(size_t)(b * N_ + row) * (2 * H_ * C_) + H_ * C_ + h * C_ + c0 + col] =
            fmaxf(val, 0.0f);
    }
}

// ---------------------------------------------------------------------------
extern "C" void kernel_launch(void* const* d_in, const int* in_sizes, int n_in,
                              void* d_out, int out_size, void* d_ws, size_t ws_size,
                              hipStream_t stream) {
    const float* X       = (const float*)d_in[0];
    const float* A       = (const float*)d_in[1];
    const float* Kern    = (const float*)d_in[2];
    const float* biases  = (const float*)d_in[3];
    const float* a_self  = (const float*)d_in[4];
    const float* a_neigh = (const float*)d_in[5];
    float* out = (float*)d_out;

    char* ws = (char*)d_ws;
    _Float16* Xh     = (_Float16*)ws;                                  // 1 MiB
    _Float16* Kt     = (_Float16*)(ws + (size_t)(1 << 20));            // 64 KiB
    float*    featsT = (float*)(ws + (size_t)(1 << 20) + (1 << 16));   // 4 MiB
    float*    as_b   = (float*)((char*)featsT + (size_t)B_ * H_ * N_ * C_ * 4);
    float*    an_b   = as_b + B_ * H_ * N_;

    k_convert_x<<<(B_ * N_ * F_ + 255) / 256, 256, 0, stream>>>(X, Xh);
    k_convert_w<<<(H_ * F_ * C_ + 255) / 256, 256, 0, stream>>>(Kern, Kt);
    k_feats<<<B_ * H_ * 16, 256, 0, stream>>>(Xh, Kt, featsT, out);
    k_logit_vec<<<B_ * H_ * N_ / 256, 256, 0, stream>>>(featsT, a_self, a_neigh, as_b, an_b);
    k_attn<<<B_ * H_ * 32, 128, 0, stream>>>(A, featsT, as_b, an_b, biases, out);
}
